// LlamaAttention_32323923870413
// MI455X (gfx1250) — compile-verified, matrix operand order checked by reading, not run on hardware
//
#include <hip/hip_runtime.h>
#include <hip/hip_bf16.h>

// ---------------------------------------------------------------------------
// LlamaAttention forward for MI455X (gfx1250, wave32, WMMA).
//
//   1) cvt_f32_bf16   : hidden, wq, wk, wv  -> bf16 (workspace)
//   2) qkv_gemm (x3)  : Y = X @ W^T  via v_wmma_f32_16x16x32_bf16,
//                       epilogue permutes to [b,h,s,d] (Q,K) / [b,h,d,s] (V^T)
//   3) qk_scores      : S = Q K^T / sqrt(128) -> f32 attn region of d_out
//   4) softmax_rows   : row softmax in place (registers + LDS reduce)
//   5) pv_gemm        : O = softmax(S) @ V, A converted f32->bf16 on the fly
//
// GEMM kernels: one wave owns a 32x64 output strip (2 M-tiles x 4 N-tiles,
// 8 f32 accumulators). The k-loop is software-pipelined with a register
// ping-pong buffer and the FINAL step peeled, so the steady-state body is an
// unconditional "issue k+32 loads, run 8 WMMAs on k" block (staged
// s_wait_loadcnt, no control-flow diamond, no prologue spills).
//
// Working sets (X+W bf16 = 67MB, Q/K/V^T bf16 = 100MB, per-head P = 16MB) fit
// in the 192MB L2, so register-blocked WMMA without LDS staging sees near-
// compulsory HBM traffic; the true HBM floor is the 1.07GB attn_weights
// stream (~185us at 23.3 TB/s).
// ---------------------------------------------------------------------------

#define HIDDEN 4096
#define NHEADS 32
#define HDIM   128
#define BATCH  2
#define SEQ    2048
#define MTOK   (BATCH * SEQ) // 4096 tokens

typedef __attribute__((ext_vector_type(16))) __bf16 v16bf;
typedef __attribute__((ext_vector_type(8)))  __bf16 v8bf;
typedef __attribute__((ext_vector_type(4)))  __bf16 v4bf;
typedef __attribute__((ext_vector_type(8)))  float  v8f;

// 16x32 bf16 A-fragment (or 32x16 B-fragment: pass N-origin as `rc`).
// Layout per ISA 7.12.2: lane&15 = row/col, lane&16 selects K-half.
static __device__ __forceinline__ v16bf frag_ld_bf16(const __bf16* __restrict__ base,
                                                     int ld, int rc, int k0, int lane) {
  const int r  = rc + (lane & 15);
  const int kb = k0 + ((lane & 16) ? 8 : 0);
  const __bf16* p = base + (size_t)r * (size_t)ld + kb;
  v8bf lo = *(const v8bf*)(p);        // K = kb .. kb+7
  v8bf hi = *(const v8bf*)(p + 16);   // K = kb+16 .. kb+23
  return __builtin_shufflevector(lo, hi, 0,1,2,3,4,5,6,7,8,9,10,11,12,13,14,15);
}

// f32 source, converted to bf16 in registers (post-wait VALU work that
// co-executes with the XDL-class WMMA pipe).
static __device__ __forceinline__ v16bf frag_ld_f32cvt(const float* __restrict__ base,
                                                       int ld, int rc, int k0, int lane) {
  const int r  = rc + (lane & 15);
  const int kb = k0 + ((lane & 16) ? 8 : 0);
  const float* p = base + (size_t)r * (size_t)ld + kb;
  v16bf out;
#pragma unroll
  for (int i = 0; i < 8; ++i) {
    out[i]     = (__bf16)p[i];
    out[i + 8] = (__bf16)p[16 + i];
  }
  return out;
}

static __device__ __forceinline__ v8f wmma_bf16(v16bf a, v16bf b, v8f c) {
  return __builtin_amdgcn_wmma_f32_16x16x32_bf16(
      /*neg_a=*/false, a, /*neg_b=*/false, b,
      /*c_mod=*/(short)0, c, /*reuse_a=*/false, /*reuse_b=*/false);
}

// ---------------------------------------------------------------------------
// 1) elementwise f32 -> bf16 (4 elems / thread)
// ---------------------------------------------------------------------------
__global__ __launch_bounds__(256) void attn_cvt_bf16_kernel(
    const float* __restrict__ in, __bf16* __restrict__ out, int n4) {
  int i = blockIdx.x * 256 + threadIdx.x;
  if (i < n4) {
    const float4 f = ((const float4*)in)[i];
    v4bf o;
    o[0] = (__bf16)f.x; o[1] = (__bf16)f.y; o[2] = (__bf16)f.z; o[3] = (__bf16)f.w;
    ((v4bf*)out)[i] = o;
  }
}

// ---------------------------------------------------------------------------
// 2) Y = X @ W^T  (M=N=K=4096); wave = 32x64 strip, pipelined + peeled k-loop.
//    W is [out,in] row-major -> B[k][n] = W[n][k] is K-contiguous per column.
// ---------------------------------------------------------------------------
__global__ __launch_bounds__(256) void attn_qkv_gemm_kernel(
    const __bf16* __restrict__ X, const __bf16* __restrict__ W,
    __bf16* __restrict__ dst, int v_transpose) {
  const int lane = threadIdx.x & 31;
  const int wid  = blockIdx.x * 8 + (threadIdx.x >> 5);
  const int m0 = (wid >> 6) << 5; // 128 M-strips of 32
  const int n0 = (wid & 63) << 6; // 64 N-strips of 64

  v8f acc[2][4] = {{v8f{}, v8f{}, v8f{}, v8f{}}, {v8f{}, v8f{}, v8f{}, v8f{}}};
  v16bf a[2][2], b[2][4];

  // prologue: fragments for k=0 into buffer 0
#pragma unroll
  for (int mi = 0; mi < 2; ++mi) a[0][mi] = frag_ld_bf16(X, HIDDEN, m0 + 16 * mi, 0, lane);
#pragma unroll
  for (int j = 0; j < 4; ++j)    b[0][j]  = frag_ld_bf16(W, HIDDEN, n0 + 16 * j,  0, lane);

  // steady state: unconditional "load k+32, compute k"
  int cur = 0;
  for (int k0 = 0; k0 < HIDDEN - 32; k0 += 32, cur ^= 1) {
    const int nk = k0 + 32;
    __builtin_prefetch(X + (size_t)(m0 + (lane & 15)) * HIDDEN + nk + 32, 0, 1);
    __builtin_prefetch(W + (size_t)(n0 + (lane & 15)) * HIDDEN + nk + 32, 0, 1);
#pragma unroll
    for (int mi = 0; mi < 2; ++mi)
      a[cur ^ 1][mi] = frag_ld_bf16(X, HIDDEN, m0 + 16 * mi, nk, lane);
#pragma unroll
    for (int j = 0; j < 4; ++j)
      b[cur ^ 1][j]  = frag_ld_bf16(W, HIDDEN, n0 + 16 * j,  nk, lane);
#pragma unroll
    for (int mi = 0; mi < 2; ++mi)
#pragma unroll
      for (int j = 0; j < 4; ++j)
        acc[mi][j] = wmma_bf16(a[cur][mi], b[cur][j], acc[mi][j]);
  }
  // peeled final k-step
#pragma unroll
  for (int mi = 0; mi < 2; ++mi)
#pragma unroll
    for (int j = 0; j < 4; ++j)
      acc[mi][j] = wmma_bf16(a[cur][mi], b[cur][j], acc[mi][j]);

#pragma unroll
  for (int mi = 0; mi < 2; ++mi)
#pragma unroll
    for (int j = 0; j < 4; ++j)
#pragma unroll
      for (int v = 0; v < 8; ++v) {
        const int row = m0 + 16 * mi + v + ((lane & 16) ? 8 : 0); // b*SEQ+s
        const int col = n0 + 16 * j + (lane & 15);                // h*128+d
        const int bb = row >> 11, s = row & (SEQ - 1);
        const int h  = col >> 7,  d = col & (HDIM - 1);
        const size_t idx = v_transpose
            ? ((size_t)(bb * NHEADS + h) * HDIM + d) * SEQ + s   // V^T [b,h,d,s]
            : ((size_t)(bb * NHEADS + h) * SEQ + s) * HDIM + d;  // Q,K [b,h,s,d]
        dst[idx] = (__bf16)acc[mi][j][v];
      }
}

// ---------------------------------------------------------------------------
// 3) S[bh] = Q[bh] @ K[bh]^T / sqrt(HDIM); wave = 32x64 strip, K-loop = 4
//    (pipelined, final step peeled; fully unrolled).
// ---------------------------------------------------------------------------
__global__ __launch_bounds__(256) void attn_qk_scores_kernel(
    const __bf16* __restrict__ Q, const __bf16* __restrict__ K,
    float* __restrict__ attn) {
  const int lane = threadIdx.x & 31;
  const int wid  = blockIdx.x * 8 + (threadIdx.x >> 5);
  const int bh  = wid >> 11;            // 64 (b,h) pairs
  const int rem = wid & 2047;
  const int q0  = (rem >> 5) << 5;      // 64 query strips of 32
  const int n0  = (rem & 31) << 6;      // 32 key strips of 64

  const __bf16* Qb = Q + (size_t)bh * SEQ * HDIM;
  const __bf16* Kb = K + (size_t)bh * SEQ * HDIM;

  v8f acc[2][4] = {{v8f{}, v8f{}, v8f{}, v8f{}}, {v8f{}, v8f{}, v8f{}, v8f{}}};
  v16bf a[2][2], b[2][4];

#pragma unroll
  for (int mi = 0; mi < 2; ++mi) a[0][mi] = frag_ld_bf16(Qb, HDIM, q0 + 16 * mi, 0, lane);
#pragma unroll
  for (int j = 0; j < 4; ++j)    b[0][j]  = frag_ld_bf16(Kb, HDIM, n0 + 16 * j,  0, lane);

  int cur = 0;
#pragma unroll
  for (int d0 = 0; d0 < HDIM - 32; d0 += 32, cur ^= 1) {
    const int nk = d0 + 32;
#pragma unroll
    for (int mi = 0; mi < 2; ++mi)
      a[cur ^ 1][mi] = frag_ld_bf16(Qb, HDIM, q0 + 16 * mi, nk, lane);
#pragma unroll
    for (int j = 0; j < 4; ++j)
      b[cur ^ 1][j]  = frag_ld_bf16(Kb, HDIM, n0 + 16 * j,  nk, lane);
#pragma unroll
    for (int mi = 0; mi < 2; ++mi)
#pragma unroll
      for (int j = 0; j < 4; ++j)
        acc[mi][j] = wmma_bf16(a[cur][mi], b[cur][j], acc[mi][j]);
  }
#pragma unroll
  for (int mi = 0; mi < 2; ++mi)
#pragma unroll
    for (int j = 0; j < 4; ++j)
      acc[mi][j] = wmma_bf16(a[cur][mi], b[cur][j], acc[mi][j]);

  const float scale = 0.08838834764831845f; // 1/sqrt(128)
  float* o = attn + (size_t)bh * SEQ * SEQ;
#pragma unroll
  for (int mi = 0; mi < 2; ++mi)
#pragma unroll
    for (int j = 0; j < 4; ++j)
#pragma unroll
      for (int v = 0; v < 8; ++v) {
        const int row = q0 + 16 * mi + v + ((lane & 16) ? 8 : 0);
        const int col = n0 + 16 * j + (lane & 15);
        o[(size_t)row * SEQ + col] = acc[mi][j][v] * scale;
      }
}

// ---------------------------------------------------------------------------
// 4) In-place row softmax (2048 elems/row); values in registers, one read +
//    one write of HBM (the bandwidth-critical 1.07GB stream).
// ---------------------------------------------------------------------------
__global__ __launch_bounds__(256) void attn_softmax_kernel(float* __restrict__ attn) {
  __shared__ float red[256];
  const int t = threadIdx.x;
  float* p = attn + (size_t)blockIdx.x * SEQ;

  float v[8];
  float mx = -3.402823466e+38f;
#pragma unroll
  for (int j = 0; j < 8; ++j) { v[j] = p[t + 256 * j]; mx = fmaxf(mx, v[j]); }
  red[t] = mx;
  __syncthreads();
  for (int s = 128; s > 0; s >>= 1) {
    if (t < s) red[t] = fmaxf(red[t], red[t + s]);
    __syncthreads();
  }
  mx = red[0];
  __syncthreads();

  float sum = 0.0f;
#pragma unroll
  for (int j = 0; j < 8; ++j) { v[j] = __expf(v[j] - mx); sum += v[j]; }
  red[t] = sum;
  __syncthreads();
  for (int s = 128; s > 0; s >>= 1) {
    if (t < s) red[t] += red[t + s];
    __syncthreads();
  }
  const float inv = 1.0f / red[0];
#pragma unroll
  for (int j = 0; j < 8; ++j) p[t + 256 * j] = v[j] * inv;
}

// ---------------------------------------------------------------------------
// 5) O[bh] = P[bh] @ V[bh]; wave = 32x64 strip, pipelined + peeled k-loop.
//    A = f32 probs (cvt in regs), B = V^T bf16 ([b,h,d,s], K-contiguous).
// ---------------------------------------------------------------------------
__global__ __launch_bounds__(256) void attn_pv_gemm_kernel(
    const float* __restrict__ P, const __bf16* __restrict__ Vt,
    float* __restrict__ out) {
  const int lane = threadIdx.x & 31;
  const int wid  = blockIdx.x * 8 + (threadIdx.x >> 5);
  const int bh  = wid >> 7;        // 64 (b,h) pairs
  const int rem = wid & 127;
  const int s0  = (rem >> 1) << 5; // 64 seq strips of 32
  const int n0  = (rem & 1) << 6;  // 2 strips of 64 over HDIM=128

  const float*  Pb = P  + (size_t)bh * SEQ * SEQ;
  const __bf16* Vb = Vt + (size_t)bh * HDIM * SEQ;

  v8f acc[2][4] = {{v8f{}, v8f{}, v8f{}, v8f{}}, {v8f{}, v8f{}, v8f{}, v8f{}}};
  v16bf a[2][2], b[2][4];

#pragma unroll
  for (int mi = 0; mi < 2; ++mi) a[0][mi] = frag_ld_f32cvt(Pb, SEQ, s0 + 16 * mi, 0, lane);
#pragma unroll
  for (int j = 0; j < 4; ++j)    b[0][j]  = frag_ld_bf16(Vb, SEQ, n0 + 16 * j,  0, lane);

  int cur = 0;
  for (int k0 = 0; k0 < SEQ - 32; k0 += 32, cur ^= 1) {
    const int nk = k0 + 32;
#pragma unroll
    for (int mi = 0; mi < 2; ++mi)
      a[cur ^ 1][mi] = frag_ld_f32cvt(Pb, SEQ, s0 + 16 * mi, nk, lane);
#pragma unroll
    for (int j = 0; j < 4; ++j)
      b[cur ^ 1][j]  = frag_ld_bf16(Vb, SEQ, n0 + 16 * j,  nk, lane);
#pragma unroll
    for (int mi = 0; mi < 2; ++mi)
#pragma unroll
      for (int j = 0; j < 4; ++j)
        acc[mi][j] = wmma_bf16(a[cur][mi], b[cur][j], acc[mi][j]);
  }
#pragma unroll
  for (int mi = 0; mi < 2; ++mi)
#pragma unroll
    for (int j = 0; j < 4; ++j)
      acc[mi][j] = wmma_bf16(a[cur][mi], b[cur][j], acc[mi][j]);

  const int bb = bh >> 5, h = bh & 31;
#pragma unroll
  for (int mi = 0; mi < 2; ++mi)
#pragma unroll
    for (int j = 0; j < 4; ++j)
#pragma unroll
      for (int v = 0; v < 8; ++v) {
        const int s = s0 + 16 * mi + v + ((lane & 16) ? 8 : 0);
        const int d = n0 + 16 * j + (lane & 15);
        out[((size_t)(bb * SEQ + s)) * HIDDEN + h * HDIM + d] = acc[mi][j][v];
      }
}

// ---------------------------------------------------------------------------
// Launch. Workspace (bf16 elems, E = 16,777,216):
//   x_bf | wq_bf | wk_bf | wv_bf | Q[b,h,s,d] | K[b,h,s,d] | V^T[b,h,d,s]
// Total 7E*2 bytes = 224 MiB.
// d_out: out (E f32) || attn_weights (16E f32).
// ---------------------------------------------------------------------------
extern "C" void kernel_launch(void* const* d_in, const int* in_sizes, int n_in,
                              void* d_out, int out_size, void* d_ws, size_t ws_size,
                              hipStream_t stream) {
  (void)in_sizes; (void)n_in; (void)out_size; (void)ws_size;

  const float* hs = (const float*)d_in[0];
  const float* wq = (const float*)d_in[1];
  const float* wk = (const float*)d_in[2];
  const float* wv = (const float*)d_in[3];

  const size_t E = (size_t)HIDDEN * (size_t)MTOK; // 16,777,216
  __bf16* x_bf  = (__bf16*)d_ws;
  __bf16* wq_bf = x_bf + E;
  __bf16* wk_bf = x_bf + 2 * E;
  __bf16* wv_bf = x_bf + 3 * E;
  __bf16* q_bf  = x_bf + 4 * E;
  __bf16* k_bf  = x_bf + 5 * E;
  __bf16* vt_bf = x_bf + 6 * E;

  float* out_o    = (float*)d_out;     // [b,s,4096]
  float* out_attn = (float*)d_out + E; // [b,h,s,s]

  // 1) convert inputs to bf16
  const int n4 = (int)(E / 4);
  const int cvt_blocks = n4 / 256;
  attn_cvt_bf16_kernel<<<cvt_blocks, 256, 0, stream>>>(hs, x_bf, n4);
  attn_cvt_bf16_kernel<<<cvt_blocks, 256, 0, stream>>>(wq, wq_bf, n4);
  attn_cvt_bf16_kernel<<<cvt_blocks, 256, 0, stream>>>(wk, wk_bf, n4);
  attn_cvt_bf16_kernel<<<cvt_blocks, 256, 0, stream>>>(wv, wv_bf, n4);

  // 2) QKV projections: (4096/32)*(4096/64) = 8192 waves -> 1024 blocks
  attn_qkv_gemm_kernel<<<1024, 256, 0, stream>>>(x_bf, wq_bf, q_bf,  0);
  attn_qkv_gemm_kernel<<<1024, 256, 0, stream>>>(x_bf, wk_bf, k_bf,  0);
  attn_qkv_gemm_kernel<<<1024, 256, 0, stream>>>(x_bf, wv_bf, vt_bf, 1);

  // 3) scores: 64 * 64 * 32 = 131072 waves -> 16384 blocks
  attn_qk_scores_kernel<<<16384, 256, 0, stream>>>(q_bf, k_bf, out_attn);

  // 4) softmax: one block per row, 2*32*2048 = 131072 rows
  attn_softmax_kernel<<<BATCH * NHEADS * SEQ, 256, 0, stream>>>(out_attn);

  // 5) P @ V: 64 * 64 * 2 = 8192 waves -> 1024 blocks
  attn_pv_gemm_kernel<<<1024, 256, 0, stream>>>(out_attn, vt_bf, out_o);
}
